// WindowAttention_609885356542
// MI455X (gfx1250) — matrix operand order repaired and checked by reading, not compiled, not run on hardware
//
#include <hip/hip_runtime.h>

// ---------------------------------------------------------------------------
// Fused Swin window attention for gfx1250 (MI455X).
// All GEMMs via V_WMMA_F32_16X16X32_BF16 (f32 accumulate, bf16 operands).
// One workgroup (8 wave32) per window; intermediates never touch HBM.
// Stage GEMMs keep 4 M-tile accumulators per wave so each weight B-fragment
// is fetched once per K-step (4x less L2 traffic, 4 independent WMMA chains).
// ---------------------------------------------------------------------------

#define DIM   256
#define NH    8
#define HD    32
#define NTOK  49
#define NP    64          // tokens padded to 64 for 16x16 tiling
#define NWIN  4096

typedef __attribute__((ext_vector_type(16))) __bf16        v16bf;
typedef __attribute__((ext_vector_type(8)))  float         v8f;
typedef __attribute__((ext_vector_type(8)))  unsigned int  v8u;

union FragU { v8u u; v16bf b; };

__device__ __forceinline__ unsigned short f32_to_bf16(float x) {
    unsigned int u = __float_as_uint(x);
    u += 0x7FFFu + ((u >> 16) & 1u);          // round-to-nearest-even
    return (unsigned short)(u >> 16);
}

// A-matrix fragment: 16x32 bf16 tile from row-major bf16 source (stride in elems).
// ISA 7.12.2: lane%16 = row; lanes<16 hold K 0..7 & 16..23, lanes>=16 hold 8..15 & 24..31.
__device__ __forceinline__ v16bf load_a(const unsigned int* p32, int stride_elems,
                                        int rbase, int kbase, int lane) {
    int row  = rbase + (lane & 15);
    int koff = (lane >> 4) << 3;
    int b    = (row * stride_elems + kbase + koff) >> 1;   // u32 index
    FragU f;
#pragma unroll
    for (int v = 0; v < 4; ++v) { f.u[v] = p32[b + v]; f.u[4 + v] = p32[b + 8 + v]; }
    return f.b;
}

// B-matrix fragment: 32x16 (KxN) from "per-column K-packed" u32 source [n][Ktot/2].
// ISA: lane%16 = column N; lanes<16 hold K 0..7 & 16..23, lanes>=16 hold 8..15 & 24..31.
__device__ __forceinline__ v16bf load_b(const unsigned int* p32, int kstride_u32,
                                        int kbase, int nbase, int lane) {
    int n    = nbase + (lane & 15);
    int koff = (lane >> 4) << 3;
    int b    = n * kstride_u32 + ((kbase + koff) >> 1);
    FragU f;
#pragma unroll
    for (int v = 0; v < 4; ++v) { f.u[v] = p32[b + v]; f.u[4 + v] = p32[b + 8 + v]; }
    return f.b;
}

#define WMMA_BF16(A, B, C) \
    __builtin_amdgcn_wmma_f32_16x16x32_bf16(false, (A), false, (B), (short)0, (C), false, false)

__device__ __forceinline__ void scatter_qk(unsigned short (*dst)[HD], const v8f& acc,
                                           int mt, int rsub, int dcol, float scale) {
#pragma unroll
    for (int j = 0; j < 8; ++j) dst[mt * 16 + rsub + j][dcol] = f32_to_bf16(acc[j] * scale);
}
__device__ __forceinline__ void scatter_vt(unsigned short (*dst)[NP], const v8f& acc,
                                           int mt, int rsub, int dcol) {
#pragma unroll
    for (int j = 0; j < 8; ++j) dst[dcol][mt * 16 + rsub + j] = f32_to_bf16(acc[j]);
}

// ---------------------------------------------------------------------------
// Kernel 0: fp32 weights -> bf16, packed [out_ch][K/2] u32 (B-fragment friendly).
// ---------------------------------------------------------------------------
__global__ void pack_weights_kernel(const float* __restrict__ qkv_w,
                                    const float* __restrict__ proj_w,
                                    unsigned int* __restrict__ wq_pack,   // [768][128]
                                    unsigned int* __restrict__ wp_pack) { // [256][128]
    int id = blockIdx.x * blockDim.x + threadIdx.x;
    const int NQ  = 768 * 128;
    const int NPJ = 256 * 128;
    if (id < NQ) {
        int o = id >> 7, kk = id & 127;
        unsigned int lo = f32_to_bf16(qkv_w[o * 256 + 2 * kk]);
        unsigned int hi = f32_to_bf16(qkv_w[o * 256 + 2 * kk + 1]);
        wq_pack[id] = lo | (hi << 16);
    } else if (id < NQ + NPJ) {
        int id2 = id - NQ;
        int o = id2 >> 7, kk = id2 & 127;
        unsigned int lo = f32_to_bf16(proj_w[o * 256 + 2 * kk]);
        unsigned int hi = f32_to_bf16(proj_w[o * 256 + 2 * kk + 1]);
        wp_pack[id2] = lo | (hi << 16);
    }
}

// ---------------------------------------------------------------------------
// Kernel 1: fused window attention. 1 block = 1 window.
// ---------------------------------------------------------------------------
__global__ __launch_bounds__(256) void win_attn_kernel(
    const float* __restrict__ x,
    const float* __restrict__ qkv_b,
    const float* __restrict__ proj_b,
    const float* __restrict__ bias_table,
    const int*   __restrict__ rel_index,
    const unsigned int* __restrict__ wq_pack,
    const unsigned int* __restrict__ wp_pack,
    float* __restrict__ out) {

    __shared__ __align__(16) unsigned short xb[NP][DIM];      // x bf16; reused as attn-out
    __shared__ __align__(16) unsigned short qs[NH][NP][HD];   // q (pre-scaled)
    __shared__ __align__(16) unsigned short ks[NH][NP][HD];   // k
    __shared__ __align__(16) unsigned short vt[NH][HD][NP];   // v transposed
    __shared__ __align__(16) float          sbuf[NP][NP];     // scores (one head)
    __shared__ __align__(16) unsigned short pb[NP][NP];       // softmax probs bf16
    __shared__ __align__(16) float          redmax[NP][4];    // softmax partial max
    __shared__ __align__(16) float          redsum[NP][4];    // softmax partial sum

    const int bwin = blockIdx.x;
    const int tid  = threadIdx.x;
    const int lane = tid & 31;
    const int w    = tid >> 5;

    // ---- load x window -> bf16 LDS (zero pad rows); zero P pad once ----
    {
        const float* xp = x + (size_t)bwin * NTOK * DIM;
        for (int i = tid; i < NP * DIM; i += 256) {
            int r = i >> 8, c = i & 255;
            xb[r][c] = f32_to_bf16(r < NTOK ? xp[r * DIM + c] : 0.f);
        }
        unsigned int* pbz = (unsigned int*)pb;
        for (int i = tid; i < NP * NP / 2; i += 256) pbz[i] = 0u;
    }
    __syncthreads();

    // ---- Stage A: qkv[64,768] = xb @ Wqkv^T + b ; scatter into q/k/vT ----
    // Each wave: 6 N-tiles; per N-tile hold 4 M-tile accumulators so each
    // weight B-fragment is loaded once per K-step.
    const unsigned int* xb32 = (const unsigned int*)xb;
#pragma unroll 1
    for (int g = 0; g < 6; ++g) {
        int nt = w + g * 8;
        float bias = qkv_b[nt * 16 + (lane & 15)];
        v8f acc0, acc1, acc2, acc3;
#pragma unroll
        for (int j = 0; j < 8; ++j) { acc0[j] = bias; acc1[j] = bias; acc2[j] = bias; acc3[j] = bias; }
#pragma unroll
        for (int kk = 0; kk < 8; ++kk) {
            v16bf bf = load_b(wq_pack, 128, kk * 32, nt * 16, lane);
            acc0 = WMMA_BF16(load_a(xb32, DIM,  0, kk * 32, lane), bf, acc0);
            acc1 = WMMA_BF16(load_a(xb32, DIM, 16, kk * 32, lane), bf, acc1);
            acc2 = WMMA_BF16(load_a(xb32, DIM, 32, kk * 32, lane), bf, acc2);
            acc3 = WMMA_BF16(load_a(xb32, DIM, 48, kk * 32, lane), bf, acc3);
        }
        int which = __builtin_amdgcn_readfirstlane(nt >> 4);        // 0=q 1=k 2=v
        int h     = __builtin_amdgcn_readfirstlane((nt & 15) >> 1);
        int dcol  = (nt & 1) * 16 + (lane & 15);
        int rsub  = (lane >> 4) << 3;
        if (which == 0) {
            scatter_qk(qs[h], acc0, 0, rsub, dcol, 0.17677669529663689f);
            scatter_qk(qs[h], acc1, 1, rsub, dcol, 0.17677669529663689f);
            scatter_qk(qs[h], acc2, 2, rsub, dcol, 0.17677669529663689f);
            scatter_qk(qs[h], acc3, 3, rsub, dcol, 0.17677669529663689f);
        } else if (which == 1) {
            scatter_qk(ks[h], acc0, 0, rsub, dcol, 1.f);
            scatter_qk(ks[h], acc1, 1, rsub, dcol, 1.f);
            scatter_qk(ks[h], acc2, 2, rsub, dcol, 1.f);
            scatter_qk(ks[h], acc3, 3, rsub, dcol, 1.f);
        } else {
            scatter_vt(vt[h], acc0, 0, rsub, dcol);
            scatter_vt(vt[h], acc1, 1, rsub, dcol);
            scatter_vt(vt[h], acc2, 2, rsub, dcol);
            scatter_vt(vt[h], acc3, 3, rsub, dcol);
        }
    }
    __syncthreads();

    // ---- per-head attention (block-cooperative, heads serialized) ----
#pragma unroll 1
    for (int h = 0; h < NH; ++h) {
        // S = q @ k^T : 4x4 tiles over 8 waves, K=32 (single WMMA per tile)
        const unsigned int* q32 = (const unsigned int*)qs[h];
        const unsigned int* k32 = (const unsigned int*)ks[h];
#pragma unroll
        for (int tt = 0; tt < 2; ++tt) {
            int t = w + tt * 8;
            int mt = t & 3, nt = t >> 2;
            v8f acc = {};
            v16bf a  = load_a(q32, HD, mt * 16, 0, lane);
            v16bf bf = load_b(k32, HD / 2, 0, nt * 16, lane);
            acc = WMMA_BF16(a, bf, acc);
            int rb  = mt * 16 + ((lane >> 4) << 3);
            int col = nt * 16 + (lane & 15);
#pragma unroll
            for (int j = 0; j < 8; ++j) sbuf[rb + j][col] = acc[j];
        }
        __syncthreads();

        // softmax over valid 49x49 region + rel-pos bias.
        // 4 threads per row; bias added in-place once; LDS tree reductions.
        {
            int r    = tid >> 2;          // 0..63
            int part = tid & 3;
            int j0   = part * 13;
            int j1   = j0 + 13; if (j1 > NTOK) j1 = NTOK;
            if (r < NTOK) {
                const int* ri = rel_index + r * NTOK;
                float mx = -1e30f;
                for (int j = j0; j < j1; ++j) {
                    float v = sbuf[r][j] + bias_table[ri[j] * NH + h];
                    sbuf[r][j] = v;
                    mx = fmaxf(mx, v);
                }
                redmax[r][part] = mx;
            }
            __syncthreads();
            if (r < NTOK) {
                float mx = fmaxf(fmaxf(redmax[r][0], redmax[r][1]),
                                 fmaxf(redmax[r][2], redmax[r][3]));
                float s = 0.f;
                for (int j = j0; j < j1; ++j) {
                    float e = __expf(sbuf[r][j] - mx);
                    sbuf[r][j] = e;
                    s += e;
                }
                redsum[r][part] = s;
            }
            __syncthreads();
            if (r < NTOK) {
                float inv = 1.f / (redsum[r][0] + redsum[r][1] + redsum[r][2] + redsum[r][3]);
                for (int j = j0; j < j1; ++j) pb[r][j] = f32_to_bf16(sbuf[r][j] * inv);
            }
        }
        __syncthreads();

        // O_h = P @ V : 4x2 tiles, one per wave, K=64 (2 WMMA steps)
        {
            const unsigned int* p32 = (const unsigned int*)pb;
            const unsigned int* v32 = (const unsigned int*)vt[h];
            int mt = w & 3, nt = w >> 2;
            v8f acc = {};
#pragma unroll
            for (int kk = 0; kk < 2; ++kk) {
                v16bf a  = load_a(p32, NP, mt * 16, kk * 32, lane);
                v16bf bf = load_b(v32, NP / 2, kk * 32, nt * 16, lane);
                acc = WMMA_BF16(a, bf, acc);
            }
            int rb  = mt * 16 + ((lane >> 4) << 3);
            int col = h * HD + nt * 16 + (lane & 15);
#pragma unroll
            for (int j = 0; j < 8; ++j) xb[rb + j][col] = f32_to_bf16(acc[j]);
        }
        __syncthreads();
    }

    // ---- projection: out = ob @ Wp^T + proj_b (rows < 49 only) ----
    const unsigned int* ob32 = (const unsigned int*)xb;
    float* op = out + (size_t)bwin * NTOK * DIM;
#pragma unroll 1
    for (int g = 0; g < 2; ++g) {
        int nt = w + g * 8;
        float bias = proj_b[nt * 16 + (lane & 15)];
        v8f acc0, acc1, acc2, acc3;
#pragma unroll
        for (int j = 0; j < 8; ++j) { acc0[j] = bias; acc1[j] = bias; acc2[j] = bias; acc3[j] = bias; }
#pragma unroll
        for (int kk = 0; kk < 8; ++kk) {
            v16bf bf = load_b(wp_pack, 128, kk * 32, nt * 16, lane);
            acc0 = WMMA_BF16(load_a(ob32, DIM,  0, kk * 32, lane), bf, acc0);
            acc1 = WMMA_BF16(load_a(ob32, DIM, 16, kk * 32, lane), bf, acc1);
            acc2 = WMMA_BF16(load_a(ob32, DIM, 32, kk * 32, lane), bf, acc2);
            acc3 = WMMA_BF16(load_a(ob32, DIM, 48, kk * 32, lane), bf, acc3);
        }
        int rsub = (lane >> 4) << 3;
        int col  = nt * 16 + (lane & 15);
#pragma unroll
        for (int mt = 0; mt < 4; ++mt) {
            const v8f& acc = (mt == 0) ? acc0 : (mt == 1) ? acc1 : (mt == 2) ? acc2 : acc3;
#pragma unroll
            for (int j = 0; j < 8; ++j) {
                int row = mt * 16 + rsub + j;
                if (row < NTOK) op[row * DIM + col] = acc[j];
            }
        }
    }
}

extern "C" void kernel_launch(void* const* d_in, const int* in_sizes, int n_in,
                              void* d_out, int out_size, void* d_ws, size_t ws_size,
                              hipStream_t stream) {
    const float* x          = (const float*)d_in[0];
    const float* qkv_w      = (const float*)d_in[1];
    const float* qkv_b      = (const float*)d_in[2];
    const float* proj_w     = (const float*)d_in[3];
    const float* proj_b     = (const float*)d_in[4];
    const float* bias_table = (const float*)d_in[5];
    const int*   rel_index  = (const int*)d_in[6];

    unsigned int* wq_pack = (unsigned int*)d_ws;           // 768*128 u32
    unsigned int* wp_pack = wq_pack + 768 * 128;           // 256*128 u32

    int total = 768 * 128 + 256 * 128;
    pack_weights_kernel<<<(total + 255) / 256, 256, 0, stream>>>(
        qkv_w, proj_w, wq_pack, wp_pack);

    win_attn_kernel<<<NWIN, 256, 0, stream>>>(
        x, qkv_b, proj_b, bias_table, rel_index, wq_pack, wp_pack, (float*)d_out);
}
